// QDenseUndirected_78142634983548
// MI455X (gfx1250) — matrix operand order repaired and checked by reading, not compiled, not run on hardware
//
#include <hip/hip_runtime.h>
#include <hip/hip_bf16.h>

typedef __attribute__((ext_vector_type(16))) _Float16 v16h;
typedef __attribute__((ext_vector_type(8)))  _Float16 v8h;
typedef __attribute__((ext_vector_type(4)))  _Float16 v4h;
typedef __attribute__((ext_vector_type(8)))  float    v8f;

#define DIM 1024
#define QDEPTH 6
#define WIRES 10
#define LDSTR 80   // LDS row stride in halves (160 B: 16B/32B aligned, bank-spread)

// ---------------------------------------------------------------------------
// Kernel A: cast x -> f16, and per-row scale = 1024 / ||x_row||^2
// (normalization is linear; fold 1/||x||^2 into the probability epilogue)
// ---------------------------------------------------------------------------
__global__ __launch_bounds__(256) void prep_kernel(const float* __restrict__ x,
                                                   _Float16* __restrict__ xh,
                                                   float* __restrict__ scale) {
  __shared__ float red[256];
  const int b = blockIdx.x;
  const int t = threadIdx.x;
  const float* row = x + (size_t)b * DIM;
  float ss = 0.f;
#pragma unroll
  for (int i = 0; i < 4; ++i) {
    float v = row[t + i * 256];
    xh[(size_t)b * DIM + t + i * 256] = (_Float16)v;
    ss += v * v;
  }
  red[t] = ss;
  __syncthreads();
  for (int o = 128; o > 0; o >>= 1) {
    if (t < o) red[t] += red[t + o];
    __syncthreads();
  }
  if (t == 0) scale[b] = 1024.0f / red[0];
}

// ---------------------------------------------------------------------------
// Kernel B: build U rows. Block n simulates the ADJOINT circuit on basis
// state e_n (state in LDS), giving U^dag e_n; conj of that is row n of U.
// Stored [n][k] contiguous-k (f16 re / f16 -im) == exactly the WMMA
// B-operand feed layout B[k,n] = U[n,k].
// ---------------------------------------------------------------------------
__device__ inline void apply_gate(float2* st, int p,
                                  float2 u00, float2 u01,
                                  float2 u10, float2 u11) {
  const unsigned mask = 1u << p;
#pragma unroll
  for (int it = 0; it < 2; ++it) {
    unsigned pr = threadIdx.x + it * 256;                        // 0..511 pair id
    unsigned i0 = ((pr & ~(mask - 1u)) << 1) | (pr & (mask - 1u));
    unsigned i1 = i0 | mask;
    float2 s0 = st[i0], s1 = st[i1];
    float2 n0, n1;
    n0.x = u00.x * s0.x - u00.y * s0.y + u01.x * s1.x - u01.y * s1.y;
    n0.y = u00.x * s0.y + u00.y * s0.x + u01.x * s1.y + u01.y * s1.x;
    n1.x = u10.x * s0.x - u10.y * s0.y + u11.x * s1.x - u11.y * s1.y;
    n1.y = u10.x * s0.y + u10.y * s0.x + u11.x * s1.y + u11.y * s1.x;
    st[i0] = n0;
    st[i1] = n1;
  }
}

__device__ inline void apply_cnot(float2* st, int pc, int pt) {
  const unsigned t = threadIdx.x;                // 8 free bits -> 256 swaps
  const int plo = pc < pt ? pc : pt;
  const int phi = pc < pt ? pt : pc;
  const unsigned mlo = 1u << plo, mhi = 1u << phi;
  unsigned i = t;
  i = ((i & ~(mlo - 1u)) << 1) | (i & (mlo - 1u));
  i = ((i & ~(mhi - 1u)) << 1) | (i & (mhi - 1u));
  i |= (1u << pc);                               // control = 1, target = 0
  unsigned j = i | (1u << pt);
  float2 a = st[i], b = st[j];
  st[i] = b;
  st[j] = a;
}

__global__ __launch_bounds__(256) void build_u_kernel(const float* __restrict__ w,
                                                      _Float16* __restrict__ ure,
                                                      _Float16* __restrict__ uim) {
  __shared__ float2 st[DIM];
  const int n = blockIdx.x;
  const int t = threadIdx.x;
#pragma unroll
  for (int i = 0; i < 4; ++i) {
    int k = t + i * 256;
    st[k] = make_float2(k == n ? 1.0f : 0.0f, 0.0f);
  }
  __syncthreads();

  // Adjoint circuit: layers in reverse; within a layer CNOTs (reversed,
  // self-inverse) then Rot-daggers (reversed).
  for (int l = QDEPTH - 1; l >= 0; --l) {
    const int r = (l % (WIRES - 1)) + 1;
    for (int q = WIRES - 1; q >= 0; --q) {
      const int tgt = (q + r) % WIRES;
      apply_cnot(st, (WIRES - 1) - q, (WIRES - 1) - tgt);
      __syncthreads();
    }
    for (int q = WIRES - 1; q >= 0; --q) {
      const float phi = tanhf(w[(l * WIRES + q) * 3 + 0]);
      const float th  = tanhf(w[(l * WIRES + q) * 3 + 1]);
      const float om  = tanhf(w[(l * WIRES + q) * 3 + 2]);
      const float c = cosf(0.5f * th), s = sinf(0.5f * th);
      const float ha = 0.5f * (phi + om), hb = 0.5f * (phi - om);
      // forward Rot entries
      float2 m00 = { cosf(ha) * c, -sinf(ha) * c };
      float2 m01 = { -cosf(hb) * s, -sinf(hb) * s };
      float2 m10 = { cosf(hb) * s, -sinf(hb) * s };
      float2 m11 = { cosf(ha) * c,  sinf(ha) * c };
      // dagger: [[conj m00, conj m10],[conj m01, conj m11]]
      float2 d00 = { m00.x, -m00.y };
      float2 d01 = { m10.x, -m10.y };
      float2 d10 = { m01.x, -m01.y };
      float2 d11 = { m11.x, -m11.y };
      apply_gate(st, (WIRES - 1) - q, d00, d01, d10, d11);
      __syncthreads();
    }
  }

  // Row n of U = conj(U^dag e_n). Pack 4 halves per 8-byte store.
  const int k0 = t * 4;
  v4h vr, vi;
#pragma unroll
  for (int j = 0; j < 4; ++j) {
    float2 a = st[k0 + j];
    vr[j] = (_Float16)a.x;
    vi[j] = (_Float16)(-a.y);
  }
  *(v4h*)&ure[(size_t)n * DIM + k0] = vr;
  *(v4h*)&uim[(size_t)n * DIM + k0] = vi;
}

// ---------------------------------------------------------------------------
// Kernel C: out[b,j] = clip(scale[b] * ((A.Ure^T)^2 + (A.Uim^T)^2), 0, 1)
// Block tile 128(M) x 64(N), 8 waves of 32x32 each, K in 64-wide chunks.
// Global->LDS staging uses the CDNA5 async DMA path
// (global_load_async_to_lds_b128, ASYNCcnt) with LDS ping-pong buffers:
// no VGPR staging (no spills), DMA overlaps the v_wmma burst.
// ---------------------------------------------------------------------------
__device__ inline v8f wmma_f16(v16h a, v16h b, v8f c) {
  return __builtin_amdgcn_wmma_f32_16x16x32_f16(
      /*neg_a=*/false, a, /*neg_b=*/false, b,
      /*c_mod=*/(short)0, c, /*reuse_a=*/false, /*reuse_b=*/false);
}

// Per-lane 16-byte global -> LDS async copy (bypasses VGPRs).
__device__ inline void async_copy_b128(const void* gptr, void* lptr) {
  unsigned lds = (unsigned)(size_t)lptr;                 // low 32b of generic = LDS byte addr
  unsigned long long ga = (unsigned long long)(size_t)gptr;
  asm volatile("global_load_async_to_lds_b128 %0, %1, off"
               :: "v"(lds), "v"(ga) : "memory");
}

__device__ inline void wait_async_all() {
  asm volatile("s_wait_asynccnt 0x0" ::: "memory");
}

__global__ __launch_bounds__(256) void gemm_probs_kernel(
    const _Float16* __restrict__ xh, const _Float16* __restrict__ ure,
    const _Float16* __restrict__ uim, const float* __restrict__ scale,
    float* __restrict__ out) {
  __shared__ _Float16 As[2][128 * LDSTR];
  __shared__ _Float16 Bre[2][64 * LDSTR];
  __shared__ _Float16 Bim[2][64 * LDSTR];

  const int bm = blockIdx.x * 128;
  const int bn = blockIdx.y * 64;
  const int tid = threadIdx.x;
  const int wave = tid >> 5;
  const int lane = tid & 31;
  const int wm = (wave >> 1) * 32;  // 0,32,64,96
  const int wn = (wave & 1) * 32;   // 0,32
  const int lm = lane & 15;
  const int kh = lane >> 4;

  // Per-thread staging geometry (fixed across K chunks):
  // A: 1024 16-byte chunks (128 rows x 8), B: 512 per matrix (64 rows x 8).
  int a_row[4], a_c8[4], b_mat[4], b_row[4], b_c8[4];
#pragma unroll
  for (int it = 0; it < 4; ++it) {
    int chunk = tid + it * 256;
    a_row[it] = chunk >> 3;
    a_c8[it] = chunk & 7;
    b_mat[it] = chunk >> 9;           // 0 = re, 1 = im
    b_row[it] = (chunk >> 3) & 63;
    b_c8[it] = chunk & 7;
  }

  v8f acc[2][2][2] = {};              // [m-sub][n-sub][re/im]

  // Issue the async DMA for one 64-wide K chunk into LDS buffer `buf`.
  auto issue_chunk = [&](int buf, int kc) {
#pragma unroll
    for (int it = 0; it < 4; ++it) {
      async_copy_b128(&xh[(size_t)(bm + a_row[it]) * DIM + kc + a_c8[it] * 8],
                      &As[buf][a_row[it] * LDSTR + a_c8[it] * 8]);
      const _Float16* src = b_mat[it] ? uim : ure;
      _Float16* dst = b_mat[it] ? &Bim[buf][0] : &Bre[buf][0];
      async_copy_b128(&src[(size_t)(bn + b_row[it]) * DIM + kc + b_c8[it] * 8],
                      &dst[b_row[it] * LDSTR + b_c8[it] * 8]);
    }
  };

  // 16 WMMAs on one 64-wide K chunk held in LDS buffer `buf`.
  auto compute_chunk = [&](int buf) {
#pragma unroll
    for (int ks = 0; ks < 64; ks += 32) {
      // A fragments: contiguous-k order, same as B -> lane m = lane&15,
      // VGPR0-3: K = kh*16 + 0..7, VGPR4-7: K = kh*16 + 8..15
      v16h afrag[2];
#pragma unroll
      for (int ms = 0; ms < 2; ++ms) {
        const _Float16* base = &As[buf][(wm + ms * 16 + lm) * LDSTR + ks + kh * 16];
        union { v16h v; v8h h[2]; } u;
        u.h[0] = *(const v8h*)base;
        u.h[1] = *(const v8h*)(base + 8);
        afrag[ms] = u.v;
      }
      // B fragments: lane n = lane&15, VGPR v: K = kh*16 + 2v,2v+1
      // -> 16 contiguous halves of the [n][k] row (aligned 32-B read)
      v16h bre[2], bim[2];
#pragma unroll
      for (int ns = 0; ns < 2; ++ns) {
        bre[ns] = *(const v16h*)&Bre[buf][(wn + ns * 16 + lm) * LDSTR + ks + kh * 16];
        bim[ns] = *(const v16h*)&Bim[buf][(wn + ns * 16 + lm) * LDSTR + ks + kh * 16];
      }
#pragma unroll
      for (int ms = 0; ms < 2; ++ms)
#pragma unroll
        for (int ns = 0; ns < 2; ++ns) {
          acc[ms][ns][0] = wmma_f16(afrag[ms], bre[ns], acc[ms][ns][0]);
          acc[ms][ns][1] = wmma_f16(afrag[ms], bim[ns], acc[ms][ns][1]);
        }
    }
  };

  // Pipeline: wait(previous DMA) + barrier -> issue next chunk's DMA ->
  // compute current chunk. Two chunks per outer step so `buf` is static.
  issue_chunk(0, 0);
  for (int kc = 0; kc < DIM; kc += 128) {
    wait_async_all();
    __syncthreads();
    if (kc + 64 < DIM) issue_chunk(1, kc + 64);
    compute_chunk(0);

    wait_async_all();
    __syncthreads();
    if (kc + 128 < DIM) issue_chunk(0, kc + 128);
    compute_chunk(1);
  }

  // Epilogue: C element (v, lane) -> M = v + 8*kh, N = lm.
  // Output is streamed once (no reuse) -> non-temporal stores keep L2 for A/B.
#pragma unroll
  for (int ms = 0; ms < 2; ++ms)
#pragma unroll
    for (int ns = 0; ns < 2; ++ns)
#pragma unroll
      for (int v = 0; v < 8; ++v) {
        int row = bm + wm + ms * 16 + kh * 8 + v;
        int col = bn + wn + ns * 16 + lm;
        float re = acc[ms][ns][0][v];
        float im = acc[ms][ns][1][v];
        float p = scale[row] * (re * re + im * im);
        p = fminf(fmaxf(p, 0.0f), 1.0f);
        __builtin_nontemporal_store(p, &out[(size_t)row * DIM + col]);
      }
}

// ---------------------------------------------------------------------------
extern "C" void kernel_launch(void* const* d_in, const int* in_sizes, int n_in,
                              void* d_out, int out_size, void* d_ws, size_t ws_size,
                              hipStream_t stream) {
  const float* x = (const float*)d_in[0];   // (4096,1,32,32) f32
  const float* w = (const float*)d_in[1];   // (6,10,3) f32
  float* out = (float*)d_out;               // (4096,1,32,32) f32

  char* ws = (char*)d_ws;
  _Float16* xh  = (_Float16*)ws;                                   // 8 MB
  _Float16* ure = (_Float16*)(ws + (size_t)8 * 1024 * 1024);       // 2 MB
  _Float16* uim = (_Float16*)(ws + (size_t)10 * 1024 * 1024);      // 2 MB
  float*    scl = (float*)(ws + (size_t)12 * 1024 * 1024);         // 16 KB

  prep_kernel<<<4096, 256, 0, stream>>>(x, xh, scl);
  build_u_kernel<<<DIM, 256, 0, stream>>>(w, ure, uim);
  gemm_probs_kernel<<<dim3(4096 / 128, DIM / 64), 256, 0, stream>>>(xh, ure, uim, scl, out);
}
